// Generator_Class_28441273434486
// MI455X (gfx1250) — compile-verified, matrix operand order checked by reading, not run on hardware
//
#include <hip/hip_runtime.h>

// ---------------------------------------------------------------------------
// Generator_Class MLP on MI455X (gfx1250, wave32)
//   prep passes fold BN-affine/ReLU + fp32->bf16 into padded bf16 tensors;
//   GEMMs are pure bf16 WMMA (128x64 block tile, 32x64 per wave) with
//   double-buffered async-to-LDS staging.
// ---------------------------------------------------------------------------

typedef __attribute__((ext_vector_type(16))) __bf16 bf16x16;
typedef __attribute__((ext_vector_type(8)))  float  f32x8;

using u16 = unsigned short;

#define CLS   16
#define BATCH 2048
#define EPSV  1e-5f
#define AST   40   // LDS row stride in ushorts (80B = 5*16B: uint4-aligned, spreads banks)

__device__ __forceinline__ u16 f2bf(float f) {
    // round-to-nearest-even fp32 -> bf16
    unsigned int u = __float_as_uint(f);
    u += 0x7FFFu + ((u >> 16) & 1u);
    return (u16)(u >> 16);
}

// --- CDNA5 async copy: global -> LDS, 16B, tracked by ASYNCcnt --------------
__device__ __forceinline__ void async_ld16(const u16* g, u16* l) {
    unsigned            lds = (unsigned)(unsigned long long)l;   // addr[31:0] = LDS offset
    unsigned long long  ga  = (unsigned long long)g;
    asm volatile("global_load_async_to_lds_b128 %0, %1, off"
                 :: "v"(lds), "v"(ga) : "memory");
}
__device__ __forceinline__ void wait_async0() {
    asm volatile("s_wait_asynccnt 0x0" ::: "memory");
}

// --- fp32 weights [C,N,Kin] -> bf16 bits [C,N,Kpad] (zero padded K) ---------
__global__ void prep_weights(const float* __restrict__ W, u16* __restrict__ Wb,
                             int N, int Kin, int Kpad) {
    size_t i = (size_t)blockIdx.x * blockDim.x + threadIdx.x;
    size_t total = (size_t)CLS * N * Kpad;
    if (i >= total) return;
    int    k  = (int)(i % Kpad);
    size_t nc = i / Kpad;                 // c*N + n
    float  v  = (k < Kin) ? W[nc * (size_t)Kin + k] : 0.0f;
    Wb[i] = f2bf(v);
}

// --- activations: fp32 [C,B,Kin] -> (BN affine + ReLU)? -> bf16 [C,B,Kpad] --
template <bool ACT>
__global__ void prep_act(const float* __restrict__ H,
                         const float* __restrict__ scale,   // [C,Kin]
                         const float* __restrict__ shift,   // [C,Kin]
                         u16* __restrict__ Ab, int Kin, int Kpad) {
    size_t i = (size_t)blockIdx.x * blockDim.x + threadIdx.x;
    size_t total = (size_t)CLS * BATCH * Kpad;
    if (i >= total) return;
    int    k  = (int)(i % Kpad);
    size_t cb = i / Kpad;                 // c*BATCH + b
    float  v  = 0.0f;
    if (k < Kin) {
        v = H[cb * (size_t)Kin + k];
        if (ACT) {
            int c = (int)(cb >> 11);      // BATCH == 2048
            v = fmaf(v, scale[c * Kin + k], shift[c * Kin + k]);
            v = fmaxf(v, 0.0f);
        }
    }
    Ab[i] = f2bf(v);
}

// --- pure bf16 WMMA GEMM: Y[c,m,n] = Ab[c,m,:] . Wb[c,n,:] + bias[c,n] ------
//     128x64 tile per 128-thread block; each wave: 32 rows x 64 cols
//     (2 A-frags share each B-frag -> 8 WMMA / 12 ds_load_b128 per K-step)
__global__ void __launch_bounds__(128)
gemm_bf16_wmma(const u16* __restrict__ Ab,     // [C,BATCH,Kpad]
               const u16* __restrict__ Wb,     // [C,N,Kpad]
               const float* __restrict__ bias, // [C,N]
               float* __restrict__ Y, int N, int Kpad) {
    __shared__ __align__(16) u16 sA[2][128 * AST];
    __shared__ __align__(16) u16 sW[2][64 * AST];

    const int c    = blockIdx.z;
    const int m0   = blockIdx.y * 128;
    const int n0   = blockIdx.x * 64;
    const int t    = threadIdx.x;
    const int lane = t & 31;
    const int wave = t >> 5;
    const int l15  = lane & 15;
    const int hsel = lane >> 4;           // 0: lanes 0-15, 1: lanes 16-31

    // staging coords: A 128 rows x 32 bf16 (1 thread/row, 4x16B);
    //                 W  64 rows x 32 bf16 (2 threads/row, 2x16B)
    const int warow = t >> 1;
    const int wkh   = (t & 1) * 16;
    const u16* gA = Ab + ((size_t)c * BATCH + m0 + t)     * (size_t)Kpad;
    const u16* gW = Wb + ((size_t)c * N    + n0 + warow)  * (size_t)Kpad + wkh;
    const int  loA = t * AST;
    const int  loW = warow * AST + wkh;

    auto issue = [&](int buf, int k0) {
        const u16* ga = gA + k0;
        u16* la = &sA[buf][loA];
        async_ld16(ga,      la);
        async_ld16(ga + 8,  la + 8);
        async_ld16(ga + 16, la + 16);
        async_ld16(ga + 24, la + 24);
        const u16* gw = gW + k0;
        u16* lw = &sW[buf][loW];
        async_ld16(gw,     lw);
        async_ld16(gw + 8, lw + 8);
    };

    f32x8 acc[8] = {};
    const int steps = Kpad >> 5;

    issue(0, 0);
    wait_async0();
    __syncthreads();

    for (int s = 0; s < steps; ++s) {
        const int cur = s & 1;
        if (s + 1 < steps) issue(1 - cur, (s + 1) << 5);   // DMA next slice during compute

        union Frag { uint4 u[2]; bf16x16 v; };
        Frag fa0, fa1;
        const u16* a0 = &sA[cur][(wave * 32 + l15) * AST + hsel * 8];
        const u16* a1 = a0 + 16 * AST;
        fa0.u[0] = *(const uint4*)(a0);        // K {0..7}  / {8..15}
        fa0.u[1] = *(const uint4*)(a0 + 16);   // K {16..23} / {24..31}
        fa1.u[0] = *(const uint4*)(a1);
        fa1.u[1] = *(const uint4*)(a1 + 16);
        #pragma unroll
        for (int q = 0; q < 4; ++q) {
            Frag fb;
            const u16* bp = &sW[cur][(q * 16 + l15) * AST + hsel * 8];
            fb.u[0] = *(const uint4*)(bp);
            fb.u[1] = *(const uint4*)(bp + 16);
            acc[q]     = __builtin_amdgcn_wmma_f32_16x16x32_bf16(
                             false, fa0.v, false, fb.v, (short)0, acc[q],     false, false);
            acc[4 + q] = __builtin_amdgcn_wmma_f32_16x16x32_bf16(
                             false, fa1.v, false, fb.v, (short)0, acc[4 + q], false, false);
        }
        if (s + 1 < steps) wait_async0();     // next buffer's DMA landed (own wave)
        __syncthreads();                      // all waves: reads done + DMA visible
    }

    // epilogue: + bias, store raw pre-BN fp32 activations
    #pragma unroll
    for (int q = 0; q < 4; ++q) {
        const int   n  = n0 + q * 16 + l15;
        const float bv = bias[c * N + n];
        #pragma unroll
        for (int half = 0; half < 2; ++half) {
            const int mb = m0 + wave * 32 + half * 16;
            #pragma unroll
            for (int i = 0; i < 8; ++i) {
                const int m = mb + (hsel ? (i + 8) : i);   // C/D VGPR layout
                Y[((size_t)c * BATCH + m) * N + n] = acc[half * 4 + q][i] + bv;
            }
        }
    }
}

// --- deterministic batch stats: sum / sumsq over m for each (c,n) ------------
__global__ void __launch_bounds__(256)
bn_stats(const float* __restrict__ Y, float* __restrict__ sum,
         float* __restrict__ sumsq, int N) {
    const int c  = blockIdx.y;
    const int n  = blockIdx.x * 64 + (threadIdx.x & 63);
    const int mg = threadIdx.x >> 6;      // 0..3
    const float* base = Y + (size_t)c * BATCH * N + n;
    float s = 0.0f, q = 0.0f;
    for (int m = mg; m < BATCH; m += 4) {
        float v = base[(size_t)m * N];
        s += v; q += v * v;
    }
    __shared__ float ss[256], qq[256];
    ss[threadIdx.x] = s; qq[threadIdx.x] = q;
    __syncthreads();
    if (mg == 0) {
        int u = threadIdx.x;
        sum  [c * N + n] = ss[u] + ss[u + 64] + ss[u + 128] + ss[u + 192];
        sumsq[c * N + n] = qq[u] + qq[u + 64] + qq[u + 128] + qq[u + 192];
    }
}

// --- fold BN into per-feature scale/shift ------------------------------------
__global__ void bn_finalize(const float* __restrict__ sum, const float* __restrict__ sumsq,
                            const float* __restrict__ g, const float* __restrict__ be,
                            float* __restrict__ scale, float* __restrict__ shift,
                            int n, float invB) {
    int i = blockIdx.x * blockDim.x + threadIdx.x;
    if (i >= n) return;
    float mu  = sum[i] * invB;
    float var = fmaxf(sumsq[i] * invB - mu * mu, 0.0f);
    float sc  = g[i] * rsqrtf(var + EPSV);
    scale[i] = sc;
    shift[i] = be[i] - mu * sc;
}

__global__ void fill_labels(float* __restrict__ out, int n, int b) {
    int i = blockIdx.x * blockDim.x + threadIdx.x;
    if (i < n) out[i] = (float)(i / b);
}

// ---------------------------------------------------------------------------
extern "C" void kernel_launch(void* const* d_in, const int* in_sizes, int n_in,
                              void* d_out, int out_size, void* d_ws, size_t ws_size,
                              hipStream_t stream) {
    (void)in_sizes; (void)n_in; (void)out_size; (void)ws_size;

    const float* x = (const float*)d_in[0];
    const float* W[5]  = {(const float*)d_in[1], (const float*)d_in[3], (const float*)d_in[5],
                          (const float*)d_in[7], (const float*)d_in[9]};
    const float* bb[5] = {(const float*)d_in[2], (const float*)d_in[4], (const float*)d_in[6],
                          (const float*)d_in[8], (const float*)d_in[10]};
    const float* g[4]  = {(const float*)d_in[11], (const float*)d_in[13],
                          (const float*)d_in[15], (const float*)d_in[17]};
    const float* be[4] = {(const float*)d_in[12], (const float*)d_in[14],
                          (const float*)d_in[16], (const float*)d_in[18]};

    static const int DIMS[6] = {100, 512, 256, 128, 256, 512};
    static const int KPAD[5] = {128, 512, 256, 128, 256};

    char*  ws  = (char*)d_ws;
    size_t off = 0;
    auto carve = [&](size_t bytes) -> char* {
        char* p = ws + off;
        off = (off + bytes + 255) & ~(size_t)255;
        return p;
    };

    u16* Wb[5];
    for (int l = 0; l < 5; ++l)
        Wb[l] = (u16*)carve((size_t)CLS * DIMS[l + 1] * KPAD[l] * sizeof(u16));
    float* hRaw   = (float*)carve((size_t)CLS * BATCH * 512 * sizeof(float));  // raw fp32 layer out
    u16*   Abf[2] = { (u16*)carve((size_t)CLS * BATCH * 512 * sizeof(u16)),
                      (u16*)carve((size_t)CLS * BATCH * 512 * sizeof(u16)) };  // bf16 padded inputs
    float* sum    = (float*)carve((size_t)CLS * 512 * sizeof(float));
    float* sumsq  = (float*)carve((size_t)CLS * 512 * sizeof(float));
    float* scale  = (float*)carve((size_t)CLS * 512 * sizeof(float));
    float* shift  = (float*)carve((size_t)CLS * 512 * sizeof(float));

    // weights -> bf16 once per call (K padded to mult. of 32)
    for (int l = 0; l < 5; ++l) {
        size_t total = (size_t)CLS * DIMS[l + 1] * KPAD[l];
        prep_weights<<<(unsigned)((total + 255) / 256), 256, 0, stream>>>(
            W[l], Wb[l], DIMS[l + 1], DIMS[l], KPAD[l]);
    }

    // x (Kin=100) -> padded bf16, no activation
    {
        size_t total = (size_t)CLS * BATCH * KPAD[0];
        prep_act<false><<<(unsigned)((total + 255) / 256), 256, 0, stream>>>(
            x, nullptr, nullptr, Abf[0], DIMS[0], KPAD[0]);
    }

    for (int l = 0; l < 5; ++l) {
        const int N = DIMS[l + 1];
        float* out = (l == 4) ? (float*)d_out : hRaw;
        dim3 grid(N / 64, BATCH / 128, CLS);
        gemm_bf16_wmma<<<grid, 128, 0, stream>>>(Abf[l & 1], Wb[l], bb[l], out, N, KPAD[l]);
        if (l < 4) {
            bn_stats<<<dim3(N / 64, CLS), 256, 0, stream>>>(hRaw, sum, sumsq, N);
            bn_finalize<<<(CLS * N + 255) / 256, 256, 0, stream>>>(
                sum, sumsq, g[l], be[l], scale, shift, CLS * N, 1.0f / BATCH);
            // next layer's input: BN+ReLU folded, bf16 (Kpad[l+1] == N)
            size_t total = (size_t)CLS * BATCH * KPAD[l + 1];
            prep_act<true><<<(unsigned)((total + 255) / 256), 256, 0, stream>>>(
                hRaw, scale, shift, Abf[(l + 1) & 1], N, KPAD[l + 1]);
        }
    }

    // genlabel after gendata (C*B*NC floats)
    float* lbl = (float*)d_out + (size_t)CLS * BATCH * 512;
    fill_labels<<<(CLS * BATCH + 255) / 256, 256, 0, stream>>>(lbl, CLS * BATCH, BATCH);
}
